// FullyConnectedTensorProduct_30176440222427
// MI455X (gfx1250) — matrix operand order repaired and checked by reading, not compiled, not run on hardware
//
#include <hip/hip_runtime.h>

typedef __attribute__((ext_vector_type(16))) __bf16       v16bf;
typedef __attribute__((ext_vector_type(8)))  float        v8f;
typedef __attribute__((ext_vector_type(8)))  unsigned int v8u;
typedef __attribute__((ext_vector_type(4)))  float        f4;

#define ROWLEN 224   // 32 (out0e) + 96 (out1o) + 96 (out1e)
#define X1LEN  128   // 32 scalars + 32*3 vector components

// bf16(a0*b0) into [15:0], bf16(a1*b1) into [31:16] — fma computed in f32, RNE
static __device__ __forceinline__ unsigned int mul2bf(float a0, float b0,
                                                      float a1, float b1) {
  unsigned int d;
  asm("v_fma_mixlo_bf16 %0, %1, %2, 0" : "=v"(d) : "v"(a0), "v"(b0));
  asm("v_fma_mixhi_bf16 %0, %2, %3, 0" : "=v"(d) : "0"(d), "v"(a1), "v"(b1));
  return d;
}
// bf16(a0*b0+c0) into [15:0], bf16(a1*b1+c1) into [31:16]
static __device__ __forceinline__ unsigned int fma2bf(float a0, float b0, float c0,
                                                      float a1, float b1, float c1) {
  unsigned int d;
  asm("v_fma_mixlo_bf16 %0, %1, %2, %3" : "=v"(d) : "v"(a0), "v"(b0), "v"(c0));
  asm("v_fma_mixhi_bf16 %0, %2, %3, %4" : "=v"(d) : "0"(d), "v"(a1), "v"(b1), "v"(c1));
  return d;
}
// bf16(a0), bf16(a1) packed (via *1.0 + 0)
static __device__ __forceinline__ unsigned int cvt2bf(float a0, float a1) {
  unsigned int d;
  asm("v_fma_mixlo_bf16 %0, %1, 1.0, 0" : "=v"(d) : "v"(a0));
  asm("v_fma_mixhi_bf16 %0, %2, 1.0, 0" : "=v"(d) : "0"(d), "v"(a1));
  return d;
}
static __device__ __forceinline__ v8f wmma_bf16(v16bf a, v16bf b, v8f c) {
  // D = A(16x32 bf16) * B(32x16 bf16) + C(16x16 f32)
  return __builtin_amdgcn_wmma_f32_16x16x32_bf16(false, a, false, b, (short)0, c, false, false);
}
static __device__ __forceinline__ f4 ntld4(const float* p) {
  return __builtin_nontemporal_load((const f4*)p);
}

__global__ __launch_bounds__(256) void fctp_kernel(
    const float* __restrict__ x1,   const float* __restrict__ x2,
    const float* __restrict__ w000, const float* __restrict__ w011,
    const float* __restrict__ w101, const float* __restrict__ w110,
    const float* __restrict__ w111,
    float* __restrict__ out, int ntiles)
{
  const int lane = threadIdx.x & 31;
  const int col  = lane & 15;        // = row-in-tile m for A, = column n for B/D
  const int h    = lane >> 4;        // lane half
  const int wave = blockIdx.x * (blockDim.x >> 5) + (threadIdx.x >> 5);
  const int nwaves = gridDim.x * (blockDim.x >> 5);

  // ---- pack the five 32x32 weight matrices as resident WMMA B operands ----
  // B element e of lane L = W[K = (e>>3)*16 + h*8 + (e&7)][N = col + 16*t], W row-major (u,w)
  v16bf Bm[5][2];
  {
    const float* Wp[5] = {w000, w011, w101, w110, w111};
    #pragma unroll
    for (int s = 0; s < 5; ++s) {
      #pragma unroll
      for (int t = 0; t < 2; ++t) {
        const float* W = Wp[s] + (col + 16 * t) + (h * 8) * 32;
        v8u pk;
        #pragma unroll
        for (int w = 0; w < 8; ++w)
          pk[w] = cvt2bf(W[((w >> 2) * 16 + 2 * (w & 3)) * 32], W[((w >> 2) * 16 + 2 * (w & 3) + 1) * 32]);
        Bm[s][t] = __builtin_bit_cast(v16bf, pk);
      }
    }
  }

  const float inv3 = 0.57735026918962576f;  // 1/sqrt(3)
  const float inv2 = 0.70710678118654752f;  // 1/sqrt(2)
  const v8f zero = {0.f, 0.f, 0.f, 0.f, 0.f, 0.f, 0.f, 0.f};

  for (int tile = wave; tile < ntiles; tile += nwaves) {
    const int row0 = tile << 4;
    const float* x1p = x1 + (size_t)(row0 + col) * X1LEN;

    // per-sample x2: s2, v2 (duplicated across the two lane halves)
    const f4 q2 = ntld4(x2 + (size_t)(row0 + col) * 4);
    const float s2 = q2[0], v2x = q2[1], v2y = q2[2], v2z = q2[3];
    const float d2x = v2x * inv3, d2y = v2y * inv3, d2z = v2z * inv3;
    const float c2x = v2x * inv2, c2y = v2y * inv2, c2z = v2z * inv2;

    // each lane loads its disjoint half-row of x1 (16B-aligned b128 loads)
    f4 sc[4], vc[12];
    #pragma unroll
    for (int g = 0; g < 2; ++g) {
      const float* sp = x1p + g * 16 + h * 8;          // s1 chunk: 8 floats
      sc[2 * g]     = ntld4(sp);
      sc[2 * g + 1] = ntld4(sp + 4);
      const float* vp = x1p + 32 + 48 * g + 24 * h;    // v1 chunk: 24 floats
      #pragma unroll
      for (int j = 0; j < 6; ++j) vc[6 * g + j] = ntld4(vp + 4 * j);
    }

    // build the 11 A operands (16x32 bf16), K index u = g*16 + h*8 + o (e = 8g + o)
    v8u pss, pdd, psx, psy, psz, pvx, pvy, pvz, pcx, pcy, pcz;
    #pragma unroll
    for (int wd = 0; wd < 8; ++wd) {
      const int e0 = 2 * wd, e1 = e0 + 1;
      const int g0 = e0 >> 3, o0 = e0 & 7, g1 = e1 >> 3, o1 = e1 & 7;
      const float s0 = sc[2 * g0 + (o0 >> 2)][o0 & 3];
      const float s1 = sc[2 * g1 + (o1 >> 2)][o1 & 3];
      const int i0 = 3 * o0, i1 = 3 * o1;
      const float vx0 = vc[6 * g0 + ((i0    ) >> 2)][(i0    ) & 3];
      const float vy0 = vc[6 * g0 + ((i0 + 1) >> 2)][(i0 + 1) & 3];
      const float vz0 = vc[6 * g0 + ((i0 + 2) >> 2)][(i0 + 2) & 3];
      const float vx1 = vc[6 * g1 + ((i1    ) >> 2)][(i1    ) & 3];
      const float vy1 = vc[6 * g1 + ((i1 + 1) >> 2)][(i1 + 1) & 3];
      const float vz1 = vc[6 * g1 + ((i1 + 2) >> 2)][(i1 + 2) & 3];

      pss[wd] = mul2bf(s0, s2, s1, s2);                    // -> W000
      psx[wd] = mul2bf(s0, v2x, s1, v2x);                  // -> W011
      psy[wd] = mul2bf(s0, v2y, s1, v2y);
      psz[wd] = mul2bf(s0, v2z, s1, v2z);
      pvx[wd] = mul2bf(vx0, s2, vx1, s2);                  // -> W101
      pvy[wd] = mul2bf(vy0, s2, vy1, s2);
      pvz[wd] = mul2bf(vz0, s2, vz1, s2);
      // dot(v1, v2)/sqrt3 -> W110 : partial in f32, last term rides the mix op
      const float p0 = vx0 * d2x + vy0 * d2y;
      const float p1 = vx1 * d2x + vy1 * d2y;
      pdd[wd] = fma2bf(vz0, d2z, p0, vz1, d2z, p1);
      // cross(v1, v2)/sqrt2 -> W111 : negated mul partial, mix op finishes
      const float mx0 = -(vz0 * c2y), mx1 = -(vz1 * c2y);
      pcx[wd] = fma2bf(vy0, c2z, mx0, vy1, c2z, mx1);
      const float my0 = -(vx0 * c2z), my1 = -(vx1 * c2z);
      pcy[wd] = fma2bf(vz0, c2x, my0, vz1, c2x, my1);
      const float mz0 = -(vy0 * c2x), mz1 = -(vy1 * c2x);
      pcz[wd] = fma2bf(vx0, c2y, mz0, vx1, c2y, mz1);
    }
    const v16bf Ass = __builtin_bit_cast(v16bf, pss);
    const v16bf Add = __builtin_bit_cast(v16bf, pdd);
    const v16bf Asx = __builtin_bit_cast(v16bf, psx);
    const v16bf Asy = __builtin_bit_cast(v16bf, psy);
    const v16bf Asz = __builtin_bit_cast(v16bf, psz);
    const v16bf Avx = __builtin_bit_cast(v16bf, pvx);
    const v16bf Avy = __builtin_bit_cast(v16bf, pvy);
    const v16bf Avz = __builtin_bit_cast(v16bf, pvz);
    const v16bf Acx = __builtin_bit_cast(v16bf, pcx);
    const v16bf Acy = __builtin_bit_cast(v16bf, pcy);
    const v16bf Acz = __builtin_bit_cast(v16bf, pcz);

    // 22 WMMAs per 16-sample tile  (Bm: 0=W000 1=W011 2=W101 3=W110 4=W111)
    v8f o0a = wmma_bf16(Add, Bm[3][0], wmma_bf16(Ass, Bm[0][0], zero));
    v8f o0b = wmma_bf16(Add, Bm[3][1], wmma_bf16(Ass, Bm[0][1], zero));
    v8f oxa = wmma_bf16(Avx, Bm[2][0], wmma_bf16(Asx, Bm[1][0], zero));
    v8f oxb = wmma_bf16(Avx, Bm[2][1], wmma_bf16(Asx, Bm[1][1], zero));
    v8f oya = wmma_bf16(Avy, Bm[2][0], wmma_bf16(Asy, Bm[1][0], zero));
    v8f oyb = wmma_bf16(Avy, Bm[2][1], wmma_bf16(Asy, Bm[1][1], zero));
    v8f oza = wmma_bf16(Avz, Bm[2][0], wmma_bf16(Asz, Bm[1][0], zero));
    v8f ozb = wmma_bf16(Avz, Bm[2][1], wmma_bf16(Asz, Bm[1][1], zero));
    v8f exa = wmma_bf16(Acx, Bm[4][0], zero);
    v8f exb = wmma_bf16(Acx, Bm[4][1], zero);
    v8f eya = wmma_bf16(Acy, Bm[4][0], zero);
    v8f eyb = wmma_bf16(Acy, Bm[4][1], zero);
    v8f eza = wmma_bf16(Acz, Bm[4][0], zero);
    v8f ezb = wmma_bf16(Acz, Bm[4][1], zero);

    // D layout: vgpr r -> row (r + 8*h), column col (+16 for second N-tile)
    #pragma unroll
    for (int r = 0; r < 8; ++r) {
      float* orow = out + (size_t)(row0 + r + 8 * h) * ROWLEN;
      __builtin_nontemporal_store(o0a[r], orow + col);
      __builtin_nontemporal_store(o0b[r], orow + col + 16);
      float* p = orow + 32 + 3 * col;
      __builtin_nontemporal_store(oxa[r], p);
      __builtin_nontemporal_store(oya[r], p + 1);
      __builtin_nontemporal_store(oza[r], p + 2);
      p = orow + 32 + 3 * (col + 16);
      __builtin_nontemporal_store(oxb[r], p);
      __builtin_nontemporal_store(oyb[r], p + 1);
      __builtin_nontemporal_store(ozb[r], p + 2);
      p = orow + 128 + 3 * col;
      __builtin_nontemporal_store(exa[r], p);
      __builtin_nontemporal_store(eya[r], p + 1);
      __builtin_nontemporal_store(eza[r], p + 2);
      p = orow + 128 + 3 * (col + 16);
      __builtin_nontemporal_store(exb[r], p);
      __builtin_nontemporal_store(eyb[r], p + 1);
      __builtin_nontemporal_store(ezb[r], p + 2);
    }
  }
}

extern "C" void kernel_launch(void* const* d_in, const int* in_sizes, int n_in,
                              void* d_out, int out_size, void* d_ws, size_t ws_size,
                              hipStream_t stream) {
  const float* x1   = (const float*)d_in[0];
  const float* x2   = (const float*)d_in[1];
  const float* w000 = (const float*)d_in[2];
  const float* w011 = (const float*)d_in[3];
  const float* w101 = (const float*)d_in[4];
  const float* w110 = (const float*)d_in[5];
  const float* w111 = (const float*)d_in[6];
  float* out = (float*)d_out;

  const int n = in_sizes[0] / X1LEN;      // 524288
  const int ntiles = n >> 4;              // 16 samples per wave-tile
  int blocks = (ntiles + 7) / 8;          // 8 waves per 256-thread block
  if (blocks > 1024) blocks = 1024;       // grid-stride: ~4 tiles/wave, weights amortized
  if (blocks < 1) blocks = 1;

  fctp_kernel<<<blocks, 256, 0, stream>>>(x1, x2, w000, w011, w101, w110, w111,
                                          out, ntiles);
}
